// Model_3332894621729
// MI455X (gfx1250) — matrix operand order repaired and checked by reading, not compiled, not run on hardware
//
#include <hip/hip_runtime.h>
#include <hip/hip_bf16.h>
#include <math.h>

typedef __attribute__((ext_vector_type(16))) _Float16 v16h;
typedef __attribute__((ext_vector_type(8)))  _Float16 v8h;
typedef __attribute__((ext_vector_type(8)))  float    v8f;

#define Tn 12

static __device__ __forceinline__ v8f wmma16(v16h a, v16h b, v8f c) {
  // D = A(16x32 f16) * B(32x16 f16) + C(16x16 f32)
  return __builtin_amdgcn_wmma_f32_16x16x32_f16(false, a, false, b, (short)0, c, false, false);
}
static __device__ __forceinline__ v16h cat16(v8h lo, v8h hi) {
  return __builtin_shufflevector(lo, hi, 0, 1, 2, 3, 4, 5, 6, 7, 8, 9, 10, 11, 12, 13, 14, 15);
}

// ---------------------------------------------------------------------------
// Kernel 0: cast to f16 with zero padding.
//   adjh  [1008][1024]  row-major (A operand: K contiguous)
//   cembT [48][1024]    transposed (B operand: K contiguous per output column)
//   nembT [48][1024]    transposed
//   wghT  [32][64]      transposed (B operand of stage-2 GEMM)
// ---------------------------------------------------------------------------
__global__ void cvt_kernel(const float* __restrict__ adj, const float* __restrict__ c_emb,
                           const float* __restrict__ n_emb, const float* __restrict__ Wg,
                           _Float16* __restrict__ adjh, _Float16* __restrict__ cembT,
                           _Float16* __restrict__ nembT, _Float16* __restrict__ wghT) {
  long i = (long)blockIdx.x * blockDim.x + threadIdx.x;
  const long NA = 1008L * 1024L;
  if (i < NA) {
    int c = (int)(i >> 10), d = (int)(i & 1023);
    adjh[i] = (_Float16)((c < 1000 && d < 1000) ? adj[c * 1000 + d] : 0.f);
    return;
  }
  i -= NA;
  if (i < 48 * 1024) {
    int s = (int)(i >> 10), d = (int)(i & 1023);
    cembT[i] = (_Float16)((d < 1000) ? c_emb[d * 48 + s] : 0.f);
    return;
  }
  i -= 48 * 1024;
  if (i < 48 * 1024) {
    int s = (int)(i >> 10), d = (int)(i & 1023);
    nembT[i] = (_Float16)((d < 1000) ? n_emb[d * 48 + s] : 0.f);
    return;
  }
  i -= 48 * 1024;
  if (i < 32 * 64) {
    int g = (int)(i >> 6), k = (int)(i & 63);
    wghT[i] = (_Float16)((k < 48) ? Wg[k * 32 + g] : 0.f);
  }
}

// ---------------------------------------------------------------------------
// Kernel 1: fused graph layer per (b,t).  Stage 1 WMMA: ac/an = (adj·mask) @ emb
// (mask folded into the A operand: 16 pk-muls / K-step instead of 48).
// Combine -> stage 2 WMMA: @Wg (K=64) -> leakyReLU -> pooled reductions.
// ---------------------------------------------------------------------------
__global__ void __launch_bounds__(256)
graph_kernel(const float* __restrict__ code_x, const float* __restrict__ divided,
             const float* __restrict__ neighbors, const int* __restrict__ lens,
             const float* __restrict__ c_emb, const float* __restrict__ n_emb,
             const float* __restrict__ u_emb, const float* __restrict__ bg,
             const _Float16* __restrict__ adjh, const _Float16* __restrict__ cembT,
             const _Float16* __restrict__ nembT, const _Float16* __restrict__ wghT,
             float* __restrict__ x_pool, float* __restrict__ em, float* __restrict__ em_f)
{
  const int bt = blockIdx.x;
  const int b = bt / Tn, t = bt % Tn;
  const int tid = threadIdx.x;
  const int lane = tid & 31, wave = tid >> 5;
  const int half = lane >> 4, n = lane & 15;

  __shared__ __align__(32) _Float16 s_cxh[1024];
  __shared__ __align__(32) _Float16 s_nbh[1024];
  __shared__ __align__(32) _Float16 s_comb[8 * 2 * 1024];   // per-wave [2][16][64] f16
  __shared__ float s_xpool[32], s_em3[32], s_emnext[32], s_emf[32];

  if (tid < 32) { s_xpool[tid] = 0.f; s_em3[tid] = 0.f; s_emnext[tid] = 0.f; s_emf[tid] = 0.f; }
  for (int d = tid; d < 1024; d += 256) {
    s_cxh[d] = (_Float16)((d < 1000) ? code_x[bt * 1000 + d] : 0.f);
    s_nbh[d] = (_Float16)((d < 1000) ? neighbors[bt * 1000 + d] : 0.f);
  }
  // zero padded K columns [48,64) of comb staging once
  for (int i = tid; i < 16 * 256; i += 256) {
    int region = i >> 8, rem = i & 255;
    s_comb[region * 1024 + (rem >> 4) * 64 + 48 + (rem & 15)] = (_Float16)0.f;
  }
  __syncthreads();

  // m3 @ u_emb contribution into em[b,t]
  for (int c = tid; c < 1000; c += 256) {
    float m3 = divided[(bt * 1000 + c) * 3 + 2];
    if (m3 != 0.f) {
      for (int g = 0; g < 32; ++g) atomicAdd(&s_em3[g], m3 * u_emb[c * 32 + g]);
    }
  }

  const int len_b = lens[b];
  const bool do_next  = (t + 1 < Tn);
  const bool do_final = (t == len_b - 1);

  _Float16* combC = &s_comb[wave * 2048];
  _Float16* combN = combC + 1024;
  const v8f z8 = {0.f, 0.f, 0.f, 0.f, 0.f, 0.f, 0.f, 0.f};

  for (int mtile = wave; mtile < 63; mtile += 8) {   // wave-uniform; EXEC all-1 at WMMA
    v8f accA[3] = {z8, z8, z8};   // ac tiles (N = 0..47)
    v8f accB[3] = {z8, z8, z8};   // an tiles
    const _Float16* arowp = adjh + (long)(mtile * 16 + n) * 1024;  // A row (M = lane&15)
    if (mtile + 8 < 63) __builtin_prefetch(arowp + 8L * 16 * 1024, 0, 1);  // next M-tile row
    for (int ks = 0; ks < 32; ++ks) {
      const int K0 = ks * 32;
      // A fragment: contiguous layout — lanes 0-15: K{0..15}, lanes 16-31: K{16..31}
      v16h a = cat16(*(const v8h*)&arowp[K0 + 16 * half],
                     *(const v8h*)&arowp[K0 + 16 * half + 8]);
      // fold 0/1 masks into A (diag(mask) commutes into adj side)
      v16h cxA = cat16(*(const v8h*)&s_cxh[K0 + 16 * half],
                       *(const v8h*)&s_cxh[K0 + 16 * half + 8]);
      v16h nbA = cat16(*(const v8h*)&s_nbh[K0 + 16 * half],
                       *(const v8h*)&s_nbh[K0 + 16 * half + 8]);
      v16h aC = a * cxA;
      v16h aN = a * nbA;
      const int kb = K0 + 16 * half;
      #pragma unroll
      for (int nt = 0; nt < 3; ++nt) {
        const int s = nt * 16 + n;
        accA[nt] = wmma16(aC, *(const v16h*)&cembT[s * 1024 + kb], accA[nt]);
        accB[nt] = wmma16(aN, *(const v16h*)&nembT[s * 1024 + kb], accB[nt]);
      }
    }
    // elementwise combine, f16 into per-wave LDS tiles (D-layout: M = r + 8*half, N = n)
    #pragma unroll
    for (int nt = 0; nt < 3; ++nt) {
      #pragma unroll
      for (int r = 0; r < 8; ++r) {
        const int m = r + 8 * half;
        const int M = mtile * 16 + m;
        const int s = nt * 16 + n;
        float cc = 0.f, nn = 0.f;
        if (M < 1000) {
          const float cx  = (float)s_cxh[M];
          const float nb  = (float)s_nbh[M];
          const float su2 = accA[nt][r] + accB[nt][r];
          cc = cx * c_emb[M * 48 + s] + cx * su2;
          nn = nb * n_emb[M * 48 + s] + nb * su2;
        }
        combC[m * 64 + s] = (_Float16)cc;
        combN[m * 64 + s] = (_Float16)nn;
      }
    }
    asm volatile("s_wait_dscnt 0" ::: "memory");   // wave-private LDS RAW fence
    __builtin_amdgcn_wave_barrier();

    // stage 2: (16x64) @ Wg(64x32)
    v8f co[2] = {z8, z8}, no_[2] = {z8, z8};
    #pragma unroll
    for (int ks2 = 0; ks2 < 2; ++ks2) {
      const int K0 = ks2 * 32;
      v16h aC = cat16(*(const v8h*)&combC[n * 64 + K0 + 16 * half],
                      *(const v8h*)&combC[n * 64 + K0 + 16 * half + 8]);
      v16h aN = cat16(*(const v8h*)&combN[n * 64 + K0 + 16 * half],
                      *(const v8h*)&combN[n * 64 + K0 + 16 * half + 8]);
      #pragma unroll
      for (int nt = 0; nt < 2; ++nt) {
        v16h bW = *(const v16h*)&wghT[(nt * 16 + n) * 64 + K0 + 16 * half];
        co[nt]  = wmma16(aC, bW, co[nt]);
        no_[nt] = wmma16(aN, bW, no_[nt]);
      }
    }

    // + bg, leakyReLU, masked poolings over the 16 c-rows of this tile
    float p_x[2]  = {0.f, 0.f};
    float p_en[2] = {0.f, 0.f};
    float p_ef[2] = {0.f, 0.f};
    #pragma unroll
    for (int nt = 0; nt < 2; ++nt) {
      const float bgv = bg[nt * 16 + n];
      #pragma unroll
      for (int r = 0; r < 8; ++r) {
        const int M = mtile * 16 + r + 8 * half;
        if (M < 1000) {
          float v1 = co[nt][r] + bgv;  v1 = (v1 > 0.f) ? v1 : 0.01f * v1;
          float v2 = no_[nt][r] + bgv; v2 = (v2 > 0.f) ? v2 : 0.01f * v2;
          p_x[nt] += divided[(bt * 1000 + M) * 3 + 0] * v1;
          if (do_next)  p_en[nt] += divided[((bt + 1) * 1000 + M) * 3 + 1] * v2;
          if (do_final) p_ef[nt] += divided[(bt * 1000 + M) * 3 + 1] * v2;
        }
      }
    }
    #pragma unroll
    for (int nt = 0; nt < 2; ++nt) {
      float vx = p_x[nt]  + __shfl_xor(p_x[nt], 16, 32);
      float ve = p_en[nt] + __shfl_xor(p_en[nt], 16, 32);
      float vf = p_ef[nt] + __shfl_xor(p_ef[nt], 16, 32);
      if (half == 0) {
        atomicAdd(&s_xpool[nt * 16 + n], vx);
        if (do_next)  atomicAdd(&s_emnext[nt * 16 + n], ve);
        if (do_final) atomicAdd(&s_emf[nt * 16 + n], vf);
      }
    }
  }
  __syncthreads();
  if (tid < 32) {
    x_pool[bt * 32 + tid] = s_xpool[tid];
    atomicAdd(&em[bt * 32 + tid], s_em3[tid]);
    if (do_next)  atomicAdd(&em[(bt + 1) * 32 + tid], s_emnext[tid]);
    if (do_final) atomicAdd(&em_f[b * 32 + tid], s_emf[tid]);
  }
}

// ---------------------------------------------------------------------------
// Kernel 2: per-batch GRU scan + final visit + attention + output GEMV
// ---------------------------------------------------------------------------
__global__ void __launch_bounds__(256)
seq_kernel(const float* __restrict__ x_pool, const float* __restrict__ em,
           const float* __restrict__ em_f, const float* __restrict__ divided,
           const int* __restrict__ lens, const float* __restrict__ intervals,
           const float* __restrict__ Wa, const float* __restrict__ ba,
           const float* __restrict__ Wo, const float* __restrict__ bo,
           const float* __restrict__ Wx, const float* __restrict__ bx,
           const float* __restrict__ Wh, const float* __restrict__ bh,
           const float* __restrict__ wt, const float* __restrict__ btb,
           const float* __restrict__ Wgate, const float* __restrict__ bgate,
           const float* __restrict__ sc, const float* __restrict__ su,
           const float* __restrict__ Wq, const float* __restrict__ bq,
           const float* __restrict__ Wsv, const float* __restrict__ bsv,
           const float* __restrict__ wphi, const float* __restrict__ bphi,
           const float* __restrict__ Wqw, const float* __restrict__ bqw,
           const float* __restrict__ Wc, const float* __restrict__ bc,
           float* __restrict__ out)
{
  const int b = blockIdx.x;
  const int tid = threadIdx.x;
  const int len = lens[b];
  const int idx = len - 1;

  __shared__ float s_h[152], s_x[32], s_em[32], s_at[32];
  __shared__ float s_gx[456], s_gh[456], s_tau[64];
  __shared__ float s_feats[13][152];
  __shared__ float s_vis[152];
  __shared__ float s_red[2];
  __shared__ float s_selfw[13], s_timew[13], s_tw[13];
  __shared__ float s_avg[152];

  if (tid < 150) s_h[tid] = 0.f;
  __syncthreads();

  for (int step = 0; step <= Tn; ++step) {      // steps 0..11 = scan, 12 = final visit
    float itv;
    if (step < Tn) {
      if (tid < 32) {
        s_x[tid]  = x_pool[(b * Tn + step) * 32 + tid];
        s_em[tid] = em[(b * Tn + step) * 32 + tid];
      }
      itv = intervals[b * Tn + step];
    } else {
      if (tid < 2) s_red[tid] = 0.f;
      __syncthreads();
      float p1 = 0.f, p3 = 0.f;
      for (int c = tid; c < 1000; c += 256) {
        const float* dv = &divided[((b * Tn + idx) * 1000 + c) * 3];
        p1 += dv[0]; p3 += dv[2];
      }
      atomicAdd(&s_red[0], p1);
      atomicAdd(&s_red[1], p3);
      __syncthreads();
      if (tid < 32) { s_x[tid] = s_red[0] * sc[tid]; s_em[tid] = em_f[b * 32 + tid] + s_red[1] * su[tid]; }
      if (tid < 150) s_h[tid] = s_feats[idx][tid];   // hidden = outs[b, idx]
      itv = intervals[b * Tn + idx];
    }
    __syncthreads();
    if (tid < 32) {                                  // tanh(em @ Wa + ba)
      float a = ba[tid];
      for (int g = 0; g < 32; ++g) a += s_em[g] * Wa[g * 32 + tid];
      s_at[tid] = tanhf(a);
    }
    __syncthreads();
    if (tid < 32) {                                  // x += at @ Wo + bo
      float a = bo[tid];
      for (int ta = 0; ta < 32; ++ta) a += s_at[ta] * Wo[ta * 32 + tid];
      s_x[tid] += a;
    }
    __syncthreads();
    for (int j = tid; j < 450; j += 256) {           // gx / gh
      float a = bx[j];
      for (int g = 0; g < 32; ++g) a += s_x[g] * Wx[g * 450 + j];
      s_gx[j] = a;
      float c = bh[j];
      for (int hh = 0; hh < 150; ++hh) c += s_h[hh] * Wh[hh * 450 + j];
      s_gh[j] = c;
    }
    if (tid < 64) s_tau[tid] = tanhf(itv * wt[tid] + btb[tid]);
    __syncthreads();
    if (tid < 150) {                                 // GRU cell + time gate
      float zg = 1.f / (1.f + expf(-(s_gx[tid] + s_gh[tid])));
      float rg = 1.f / (1.f + expf(-(s_gx[150 + tid] + s_gh[150 + tid])));
      float ng = tanhf(s_gx[300 + tid] + rg * s_gh[300 + tid]);
      float hnew = (1.f - zg) * ng + zg * s_h[tid];
      float gacc = bgate[tid];
      for (int k = 0; k < 64; ++k) gacc += s_tau[k] * Wgate[k * 150 + tid];
      float o = (1.f / (1.f + expf(-gacc))) * tanhf(hnew);
      if (step < Tn) s_feats[step][tid] = o; else s_vis[tid] = o;
      s_h[tid] = hnew;
    }
    __syncthreads();
  }

  // masked feats [13][150] + visit feature at row len
  for (int i = tid; i < 13 * 150; i += 256) {
    int row = i / 150, hh = i % 150;
    float v = (row < Tn && row < len) ? s_feats[row][hh] : 0.f;
    if (row == len) v += s_vis[hh];
    s_feats[row][hh] = v;
  }
  __syncthreads();
  if (tid < 13) {
    int row = tid;
    float itvr = (row < Tn) ? intervals[b * Tn + row] : 0.f;
    float ss = bsv[0];
    for (int hh = 0; hh < 150; ++hh) ss += s_feats[row][hh] * Wsv[hh];
    float ts = 0.f;
    for (int q = 0; q < 64; ++q) {
      float qv = bq[q];
      if (row == len) for (int hh = 0; hh < 150; ++hh) qv += s_feats[row][hh] * Wq[hh * 64 + q];
      qv = fmaxf(qv, 0.f);
      ts += cosf(itvr * wphi[q] + bphi[q]) * qv;
    }
    bool masked = (row >= len);
    s_selfw[row] = masked ? -1e9f : ss;
    s_timew[row] = masked ? -1e9f : ts;
  }
  __syncthreads();
  if (tid == 0) {
    for (int pass = 0; pass < 2; ++pass) {
      float* arr = pass ? s_timew : s_selfw;
      float mx = -1e30f;
      for (int r = 0; r < 13; ++r) mx = fmaxf(mx, arr[r]);
      float sm = 0.f;
      for (int r = 0; r < 13; ++r) { arr[r] = expf(arr[r] - mx); sm += arr[r]; }
      float inv = 1.f / sm;
      for (int r = 0; r < 13; ++r) arr[r] *= inv;
    }
  }
  __syncthreads();
  if (tid < 13) {
    int row = tid;
    float l0 = bqw[0], l1 = bqw[1];
    if (row == len)
      for (int hh = 0; hh < 150; ++hh) { l0 += s_feats[row][hh] * Wqw[hh * 2]; l1 += s_feats[row][hh] * Wqw[hh * 2 + 1]; }
    float mx = fmaxf(l0, l1);
    float e0 = expf(l0 - mx), e1 = expf(l1 - mx);
    float inv = 1.f / (e0 + e1);
    s_tw[row] = s_timew[row] * (e0 * inv) + s_selfw[row] * (e1 * inv);
  }
  __syncthreads();
  if (tid == 0) {
    float sm = 0.f;
    for (int r = 0; r < 13; ++r) sm += s_tw[r];
    float inv = 1.f / (sm + 1e-5f);
    for (int r = 0; r < 13; ++r) s_tw[r] *= inv;
  }
  __syncthreads();
  if (tid < 150) {
    float a = 0.f;
    for (int r = 0; r < 13; ++r) a += s_feats[r][tid] * s_tw[r];
    s_avg[tid] = a;
  }
  __syncthreads();
  for (int o = tid; o < 1000; o += 256) {
    float a = bc[o];
    for (int hh = 0; hh < 150; ++hh) a += s_avg[hh] * Wc[hh * 1000 + o];
    out[b * 1000 + o] = a;
  }
}

// ---------------------------------------------------------------------------
extern "C" void kernel_launch(void* const* d_in, const int* in_sizes, int n_in,
                              void* d_out, int out_size, void* d_ws, size_t ws_size,
                              hipStream_t stream) {
  const float* code_x    = (const float*)d_in[0];
  const float* divided   = (const float*)d_in[1];
  const float* neighbors = (const float*)d_in[2];
  const int*   lens      = (const int*)  d_in[3];
  const float* intervals = (const float*)d_in[4];
  const float* adj   = (const float*)d_in[5];
  const float* c_emb = (const float*)d_in[6];
  const float* n_emb = (const float*)d_in[7];
  const float* u_emb = (const float*)d_in[8];
  const float* Wg    = (const float*)d_in[9];
  const float* bg    = (const float*)d_in[10];
  const float* Wa    = (const float*)d_in[11];
  const float* ba    = (const float*)d_in[12];
  const float* Wo    = (const float*)d_in[13];
  const float* bo    = (const float*)d_in[14];
  const float* Wx    = (const float*)d_in[15];
  const float* bx    = (const float*)d_in[16];
  const float* Wh    = (const float*)d_in[17];
  const float* bh    = (const float*)d_in[18];
  const float* wt    = (const float*)d_in[19];
  const float* bt    = (const float*)d_in[20];
  const float* Wgate = (const float*)d_in[21];
  const float* bgate = (const float*)d_in[22];
  const float* sc    = (const float*)d_in[23];
  const float* su    = (const float*)d_in[24];
  const float* Wq    = (const float*)d_in[25];
  const float* bq    = (const float*)d_in[26];
  const float* Wsv   = (const float*)d_in[27];
  const float* bsv   = (const float*)d_in[28];
  const float* wphi  = (const float*)d_in[29];
  const float* bphi  = (const float*)d_in[30];
  const float* Wqw   = (const float*)d_in[31];
  const float* bqw   = (const float*)d_in[32];
  const float* Wc    = (const float*)d_in[33];
  const float* bc    = (const float*)d_in[34];

  char* ws = (char*)d_ws;
  size_t off = 0;
  _Float16* adjh  = (_Float16*)(ws + off); off += 1008L * 1024 * 2;
  _Float16* cembT = (_Float16*)(ws + off); off += 48L * 1024 * 2;
  _Float16* nembT = (_Float16*)(ws + off); off += 48L * 1024 * 2;
  _Float16* wghT  = (_Float16*)(ws + off); off += 32L * 64 * 2;
  size_t zoff = off;
  float* x_pool = (float*)(ws + off); off += 96L * 32 * 4;
  float* emb    = (float*)(ws + off); off += 96L * 32 * 4;
  float* emf    = (float*)(ws + off); off += 8L * 32 * 4;

  hipMemsetAsync(ws + zoff, 0, off - zoff, stream);   // atomic accumulators -> 0

  long total = 1008L * 1024 + 2L * 48 * 1024 + 32L * 64;
  cvt_kernel<<<(int)((total + 255) / 256), 256, 0, stream>>>(
      adj, c_emb, n_emb, Wg, adjh, cembT, nembT, wghT);

  graph_kernel<<<96, 256, 0, stream>>>(
      code_x, divided, neighbors, lens, c_emb, n_emb, u_emb, bg,
      adjh, cembT, nembT, wghT, x_pool, emb, emf);

  seq_kernel<<<8, 256, 0, stream>>>(
      x_pool, emb, emf, divided, lens, intervals,
      Wa, ba, Wo, bo, Wx, bx, Wh, bh, wt, bt, Wgate, bgate,
      sc, su, Wq, bq, Wsv, bsv, wphi, bphi, Wqw, bqw, Wc, bc,
      (float*)d_out);

  (void)in_sizes; (void)n_in; (void)out_size; (void)ws_size;
}
